// CameraNeRF_7009386627336
// MI455X (gfx1250) — matrix operand order repaired and checked by reading, not compiled, not run on hardware
//
#include <hip/hip_runtime.h>
#include <hip/hip_bf16.h>
#include <math.h>

typedef _Float16 half_t;
typedef __attribute__((ext_vector_type(16))) _Float16 v16h;
typedef __attribute__((ext_vector_type(8)))  float    v8f;
typedef __attribute__((ext_vector_type(4)))  float    v4f;

#define N_RAYS 16384
#define KK 961   // 31*31

// ---------------------------------------------------------------------------
// Weight conversion: f32 W[out x in] (row-major)  ->  f16 W^T tiles in the
// WMMA B-matrix (32x16, 16-bit) register layout:
//   tile (kb, nb), lane l: column n = nb*16 + (l&15)
//   halves e=0..15 hold K = kb*32 + e + ((e&8)?8:0) + ((l&16)?8:0)
//   storage: dst[(((kb*Nt)+nb)*32 + l)*16 + e]   (each lane's 16 halves contiguous)
// ---------------------------------------------------------------------------
__global__ void convert_weight_kernel(const float* __restrict__ W, half_t* __restrict__ dst,
                                      int out, int in, int Kpad, int NoutPad)
{
    int Nt = NoutPad >> 4;
    int Kt = Kpad >> 5;
    int gid = blockIdx.x * blockDim.x + threadIdx.x;
    int total = Kt * Nt * 32;
    if (gid >= total) return;
    int lane = gid & 31;
    int tile = gid >> 5;
    int nb = tile % Nt;
    int kb = tile / Nt;
    int n = nb * 16 + (lane & 15);
    int kBase = kb * 32 + ((lane & 16) ? 8 : 0);
    half_t* d = dst + (size_t)gid * 16;
#pragma unroll
    for (int e = 0; e < 16; ++e) {
        int k = kBase + e + ((e & 8) ? 8 : 0);
        float v = (k < in && n < out) ? W[(size_t)n * in + k] : 0.0f;
        d[e] = (half_t)v;
    }
}

// ---------------------------------------------------------------------------
// WMMA GEMM:  Y[N x out] = act( X[N x Kpad(f32)] @ W^T + b )
// Block = 128 threads (4 waves). Wave w computes rows [bx*16, +16) and exactly
// NT column tiles starting at (blockIdx.y*4 + w) * (NT*16). Grid is sized so
// every non-returning wave owns NT *full* tiles -> zero branches around WMMA,
// accumulators stay pinned in VGPRs.
// A fragment (16x32 f16) built on the fly from f32 rows per the ISA layout:
//   lanes 0-15: halves 0..7 = K 0..7,  halves 8..15 = K 16..23
//   lanes16-31: halves 0..7 = K 8..15, halves 8..15 = K 24..31
// ---------------------------------------------------------------------------
template <int NT>
__global__ __launch_bounds__(128) void gemm_wmma_kernel(
    const float* __restrict__ X, int ldx,
    const half_t* __restrict__ Bw,
    const float* __restrict__ bias,
    float* __restrict__ Y, int ldy,
    int Kpad, int Nout, int NoutPad, int doRelu)
{
    const int lane = threadIdx.x & 31;
    const int wave = threadIdx.x >> 5;
    const int rowBase = blockIdx.x << 4;
    const int colBase = (blockIdx.y * 4 + wave) * (NT * 16);
    if (colBase >= NoutPad) return;        // wave-uniform, before any WMMA
    const int Nt = NoutPad >> 4;
    const int Kt = Kpad >> 5;
    const int mlo = lane & 15;
    const float* xrow = X + (size_t)(rowBase + mlo) * ldx;
    const int loOff = (lane & 16) ? 8 : 0;
    const int hiOff = (lane & 16) ? 24 : 16;
    const int nb0 = colBase >> 4;

    v8f acc[NT];
#pragma unroll
    for (int t = 0; t < NT; ++t) acc[t] = (v8f){0.f, 0.f, 0.f, 0.f, 0.f, 0.f, 0.f, 0.f};

    for (int kb = 0; kb < Kt; ++kb) {
        const float* xk = xrow + (kb << 5);
        v4f f0 = *(const v4f*)(xk + loOff);
        v4f f1 = *(const v4f*)(xk + loOff + 4);
        v4f f2 = *(const v4f*)(xk + hiOff);
        v4f f3 = *(const v4f*)(xk + hiOff + 4);
        v16h a;
#pragma unroll
        for (int i = 0; i < 4; ++i) {
            a[i]      = (half_t)f0[i];
            a[4 + i]  = (half_t)f1[i];
            a[8 + i]  = (half_t)f2[i];
            a[12 + i] = (half_t)f3[i];
        }
        const half_t* brow = Bw + (((size_t)kb * Nt + nb0) * 32 + lane) * 16;
#pragma unroll
        for (int t = 0; t < NT; ++t) {
            v16h b = *(const v16h*)(brow + (size_t)t * 512);
            acc[t] = __builtin_amdgcn_wmma_f32_16x16x32_f16(
                false, a, false, b, (short)0, acc[t], false, false);
        }
    }

    // C/D layout: VGPR r -> M = r + ((lane&16)?8:0); N = lane&15
    const int mAdd = (lane & 16) ? 8 : 0;
#pragma unroll
    for (int t = 0; t < NT; ++t) {
        int n = colBase + t * 16 + mlo;
        if (n < Nout) {
            float bv = bias[n];
#pragma unroll
            for (int r = 0; r < 8; ++r) {
                float v = acc[t][r] + bv;
                if (doRelu) v = fmaxf(v, 0.0f);
                Y[(size_t)(rowBase + mAdd + r) * ldy + n] = v;
            }
        }
    }
}

// ---------------------------------------------------------------------------
// Write columns [c0, cEnd) of buf: first w cols from src (N x w), rest zero.
// ---------------------------------------------------------------------------
__global__ void append_cols_kernel(float* __restrict__ buf, int ld,
                                   const float* __restrict__ src, int w,
                                   int c0, int cEnd)
{
    int ray = blockIdx.x;
    for (int j = threadIdx.x; j < cEnd - c0; j += blockDim.x) {
        float v = (j < w) ? src[(size_t)ray * w + j] : 0.0f;
        buf[(size_t)ray * ld + c0 + j] = v;
    }
}

__device__ __forceinline__ float sigm(float x) { return 1.0f / (1.0f + __expf(-x)); }

// ---------------------------------------------------------------------------
// Blur 1: axis-aligned gaussian from sig[:,0:2]
// ROI layout: AIF_padding_ROIs[img][py][px][3][31][31] -> per-ray contiguous 3*961 f32
// ---------------------------------------------------------------------------
__global__ __launch_bounds__(128) void blur_simple_kernel(
    const float* __restrict__ sig, int lds,
    const int* __restrict__ idx, const int* __restrict__ pix,
    const float* __restrict__ roiBase, float* __restrict__ out)
{
    int ray = blockIdx.x;
    float s1 = 5.0f * sigm(sig[(size_t)ray * lds + 0]) + 1e-4f;
    float s2 = 5.0f * sigm(sig[(size_t)ray * lds + 1]) + 1e-4f;
    float a = 1.0f / (6.2831853071795864f * s1 * s2);
    float i1 = 1.0f / (s1 * s1), i2 = 1.0f / (s2 * s2);
    int im = idx[ray];
    int px = pix[(size_t)ray * 2 + 0], py = pix[(size_t)ray * 2 + 1];
    const float* roi = roiBase + (((size_t)im * 64 + py) * 64 + px) * (3 * KK);
    float c0 = 0, c1 = 0, c2 = 0, den = 0;
    for (int t = threadIdx.x; t < KK; t += 128) {
        __builtin_prefetch(roi + t + 256, 0, 1);
        float Xg = (float)(t / 31) - 15.0f;
        float Yg = (float)(t % 31) - 15.0f;
        float g = a * __expf(-0.5f * (Xg * Xg * i1 + Yg * Yg * i2));
        den += g;
        c0 += roi[t] * g;
        c1 += roi[KK + t] * g;
        c2 += roi[2 * KK + t] * g;
    }
    __shared__ float red[4][128];
    red[0][threadIdx.x] = c0; red[1][threadIdx.x] = c1;
    red[2][threadIdx.x] = c2; red[3][threadIdx.x] = den;
    __syncthreads();
    for (int s = 64; s > 0; s >>= 1) {
        if (threadIdx.x < (unsigned)s)
            for (int q = 0; q < 4; ++q) red[q][threadIdx.x] += red[q][threadIdx.x + s];
        __syncthreads();
    }
    if (threadIdx.x == 0) {
        float d = red[3][0] + 1e-5f;
        out[(size_t)ray * 3 + 0] = red[0][0] / d;
        out[(size_t)ray * 3 + 1] = red[1][0] / d;
        out[(size_t)ray * 3 + 2] = red[2][0] / d;
    }
}

// ---------------------------------------------------------------------------
// Blur 2: full bivariate gaussian from murho[:,0:5]
// ---------------------------------------------------------------------------
__global__ __launch_bounds__(128) void blur_mu_kernel(
    const float* __restrict__ mur, int ldm,
    const int* __restrict__ idx, const int* __restrict__ pix,
    const float* __restrict__ roiBase, float* __restrict__ out)
{
    int ray = blockIdx.x;
    const float* m = mur + (size_t)ray * ldm;
    float s1 = 5.0f * sigm(m[0]) + 1e-4f;
    float s2 = 5.0f * sigm(m[1]) + 1e-4f;
    float mu1 = m[2] + 1e-4f;
    float mu2 = m[3] + 1e-4f;
    float rho = sigm(m[4]) * 0.1f;
    float omr = 1.0f - rho * rho;
    float a = 1.0f / (6.2831853071795864f * s1 * s2 * sqrtf(omr));
    float coef = -0.5f / omr;
    float i1 = 1.0f / (s1 * s1), i2 = 1.0f / (s2 * s2), i12 = rho / (s1 * s2);
    int im = idx[ray];
    int px = pix[(size_t)ray * 2 + 0], py = pix[(size_t)ray * 2 + 1];
    const float* roi = roiBase + (((size_t)im * 64 + py) * 64 + px) * (3 * KK);
    float c0 = 0, c1 = 0, c2 = 0, den = 0;
    for (int t = threadIdx.x; t < KK; t += 128) {
        __builtin_prefetch(roi + t + 256, 0, 1);
        float dx = ((float)(t / 31) - 15.0f) - mu1;
        float dy = ((float)(t % 31) - 15.0f) - mu2;
        float g = a * __expf(coef * (dx * dx * i1 - 2.0f * i12 * dx * dy + dy * dy * i2));
        den += g;
        c0 += roi[t] * g;
        c1 += roi[KK + t] * g;
        c2 += roi[2 * KK + t] * g;
    }
    __shared__ float red[4][128];
    red[0][threadIdx.x] = c0; red[1][threadIdx.x] = c1;
    red[2][threadIdx.x] = c2; red[3][threadIdx.x] = den;
    __syncthreads();
    for (int s = 64; s > 0; s >>= 1) {
        if (threadIdx.x < (unsigned)s)
            for (int q = 0; q < 4; ++q) red[q][threadIdx.x] += red[q][threadIdx.x + s];
        __syncthreads();
    }
    if (threadIdx.x == 0) {
        float d = red[3][0] + 1e-5f;
        out[(size_t)ray * 3 + 0] = red[0][0] / d;
        out[(size_t)ray * 3 + 1] = red[1][0] / d;
        out[(size_t)ray * 3 + 2] = red[2][0] / d;
    }
}

// ---------------------------------------------------------------------------
// Blur 3: learned 961-tap kernel from wBuf (N x ldw, 961 valid cols)
// ---------------------------------------------------------------------------
__global__ __launch_bounds__(128) void blur_w_kernel(
    const float* __restrict__ wBuf, int ldw,
    const int* __restrict__ idx, const int* __restrict__ pix,
    const float* __restrict__ roiBase, float* __restrict__ out)
{
    int ray = blockIdx.x;
    const float* wrow = wBuf + (size_t)ray * ldw;
    int im = idx[ray];
    int px = pix[(size_t)ray * 2 + 0], py = pix[(size_t)ray * 2 + 1];
    const float* roi = roiBase + (((size_t)im * 64 + py) * 64 + px) * (3 * KK);
    float c0 = 0, c1 = 0, c2 = 0, den = 0;
    for (int t = threadIdx.x; t < KK; t += 128) {
        __builtin_prefetch(roi + t + 256, 0, 1);
        float wv = wrow[t];
        den += wv;
        c0 += roi[t] * wv;
        c1 += roi[KK + t] * wv;
        c2 += roi[2 * KK + t] * wv;
    }
    __shared__ float red[4][128];
    red[0][threadIdx.x] = c0; red[1][threadIdx.x] = c1;
    red[2][threadIdx.x] = c2; red[3][threadIdx.x] = den;
    __syncthreads();
    for (int s = 64; s > 0; s >>= 1) {
        if (threadIdx.x < (unsigned)s)
            for (int q = 0; q < 4; ++q) red[q][threadIdx.x] += red[q][threadIdx.x + s];
        __syncthreads();
    }
    if (threadIdx.x == 0) {
        float d = red[3][0] + 1e-5f;
        out[(size_t)ray * 3 + 0] = red[0][0] / d;
        out[(size_t)ray * 3 + 1] = red[1][0] / d;
        out[(size_t)ray * 3 + 2] = red[2][0] / d;
    }
}

__global__ void paras_kernel(const float* __restrict__ mur, int ldm,
                             float* __restrict__ out, int n)
{
    int i = blockIdx.x * blockDim.x + threadIdx.x;
    if (i >= n) return;
    const float* m = mur + (size_t)i * ldm;
    out[(size_t)i * 5 + 0] = 5.0f * sigm(m[0]) + 1e-4f;
    out[(size_t)i * 5 + 1] = 5.0f * sigm(m[1]) + 1e-4f;
    out[(size_t)i * 5 + 2] = m[2] + 1e-4f;
    out[(size_t)i * 5 + 3] = m[3] + 1e-4f;
    out[(size_t)i * 5 + 4] = sigm(m[4]) * 0.1f;
}

// ---------------------------------------------------------------------------
// Host side
// ---------------------------------------------------------------------------
static inline int padN(int out) {
    int p = (out + 15) & ~15;
    if (p > 128) p = (p + 255) & ~255;   // NT=4 tiling needs multiples of 256
    return p;                            // {16, 128, 256, 512, 1024}
}

extern "C" void kernel_launch(void* const* d_in, const int* in_sizes, int n_in,
                              void* d_out, int out_size, void* d_ws, size_t ws_size,
                              hipStream_t stream)
{
    (void)in_sizes; (void)n_in; (void)out_size; (void)ws_size;
    const int N = N_RAYS;

    const int*   idx   = (const int*)d_in[1];
    const int*   pix   = (const int*)d_in[2];
    const float* depth = (const float*)d_in[3];
    const float* fd    = (const float*)d_in[4];
    const float* imgC  = (const float*)d_in[5];
    const float* rpos  = (const float*)d_in[6];
    const float* roi   = (const float*)d_in[7];
    float* out = (float*)d_out;

    // params flattened after the 8 array inputs, sorted layer names, W then b:
    // h1 h2 l1_1 l1_2 l1_3a l1_3b l2_1 l2_2 l2_3a l2_3b l3_1 l3_2 mur_a mur_b sig_a sig_b w_a w_b
    const int PB = 8;
    struct WSpec { const float* W; const float* b; int out; int in; half_t* dev; };
    WSpec S[18] = {
        { (const float*)d_in[PB + 0],  (const float*)d_in[PB + 1],  256, 256, nullptr }, // 0  h1
        { (const float*)d_in[PB + 2],  (const float*)d_in[PB + 3],  256, 256, nullptr }, // 1  h2
        { (const float*)d_in[PB + 4],  (const float*)d_in[PB + 5],  256, 1,   nullptr }, // 2  l1_1
        { (const float*)d_in[PB + 6],  (const float*)d_in[PB + 7],  256, 257, nullptr }, // 3  l1_2
        { (const float*)d_in[PB + 8],  (const float*)d_in[PB + 9],  256, 257, nullptr }, // 4  l1_3a
        { (const float*)d_in[PB + 10], (const float*)d_in[PB + 11], 256, 256, nullptr }, // 5  l1_3b
        { (const float*)d_in[PB + 12], (const float*)d_in[PB + 13], 256, 259, nullptr }, // 6  l2_1
        { (const float*)d_in[PB + 14], (const float*)d_in[PB + 15], 256, 259, nullptr }, // 7  l2_2
        { (const float*)d_in[PB + 16], (const float*)d_in[PB + 17], 256, 258, nullptr }, // 8  l2_3a
        { (const float*)d_in[PB + 18], (const float*)d_in[PB + 19], 256, 256, nullptr }, // 9  l2_3b
        { (const float*)d_in[PB + 20], (const float*)d_in[PB + 21], 256, 256, nullptr }, // 10 l3_1
        { (const float*)d_in[PB + 22], (const float*)d_in[PB + 23], 256, 259, nullptr }, // 11 l3_2
        { (const float*)d_in[PB + 24], (const float*)d_in[PB + 25], 128, 256, nullptr }, // 12 mur_a
        { (const float*)d_in[PB + 26], (const float*)d_in[PB + 27], 5,   128, nullptr }, // 13 mur_b
        { (const float*)d_in[PB + 28], (const float*)d_in[PB + 29], 128, 256, nullptr }, // 14 sig_a
        { (const float*)d_in[PB + 30], (const float*)d_in[PB + 31], 2,   128, nullptr }, // 15 sig_b
        { (const float*)d_in[PB + 32], (const float*)d_in[PB + 33], 512, 256, nullptr }, // 16 w_a
        { (const float*)d_in[PB + 34], (const float*)d_in[PB + 35], 961, 512, nullptr }, // 17 w_b
    };

    char* ws = (char*)d_ws;
    size_t cur = 0;
    auto carve = [&](size_t bytes) -> char* {
        char* p = ws + cur;
        cur = (cur + bytes + 255) & ~(size_t)255;
        return p;
    };

    const int LD = 288;  // activation stride: 9*32, holds up to 259-wide concat + zero pad
    float* actA  = (float*)carve((size_t)N * LD * 4);
    float* actB  = (float*)carve((size_t)N * LD * 4);
    float* actXs = (float*)carve((size_t)N * LD * 4);  // x (after l1_3b), then [x, imgC]
    float* actX1 = (float*)carve((size_t)N * LD * 4);  // x1 (after l2_3b)
    float* sigB  = (float*)carve((size_t)N * 16 * 4);
    float* murB  = (float*)carve((size_t)N * 16 * 4);
    float* waB   = (float*)carve((size_t)N * 512 * 4);
    float* wB    = (float*)carve((size_t)N * 1024 * 4); // w_b output, padded to 1024

    // one-time weight conversion into WMMA-B layout
    for (int i = 0; i < 18; ++i) {
        int Kpad = (S[i].in + 31) & ~31;
        int Npad = padN(S[i].out);
        S[i].dev = (half_t*)carve((size_t)Kpad * Npad * sizeof(half_t));
        int total = (Kpad / 32) * (Npad / 16) * 32;
        convert_weight_kernel<<<(total + 255) / 256, 256, 0, stream>>>(
            S[i].W, S[i].dev, S[i].out, S[i].in, Kpad, Npad);
    }

    auto gemm = [&](const float* Xp, int ldx, int si, float* Yp, int ldy, int relu) {
        int Kpad = (S[si].in + 31) & ~31;
        int Npad = padN(S[si].out);
        if (Npad >= 256) {          // NT=4: block covers 256 cols exactly
            dim3 grid(N / 16, Npad / 256);
            gemm_wmma_kernel<4><<<grid, 128, 0, stream>>>(
                Xp, ldx, S[si].dev, S[si].b, Yp, ldy, Kpad, S[si].out, Npad, relu);
        } else if (Npad == 128) {   // NT=2: block covers 128 cols exactly
            dim3 grid(N / 16, 1);
            gemm_wmma_kernel<2><<<grid, 128, 0, stream>>>(
                Xp, ldx, S[si].dev, S[si].b, Yp, ldy, Kpad, S[si].out, Npad, relu);
        } else {                    // NT=1: wave0 owns the single 16-col tile
            dim3 grid(N / 16, 1);
            gemm_wmma_kernel<1><<<grid, 128, 0, stream>>>(
                Xp, ldx, S[si].dev, S[si].b, Yp, ldy, Kpad, S[si].out, Npad, relu);
        }
    };
    auto append = [&](float* buf, const float* src, int w, int c0, int cEnd) {
        append_cols_kernel<<<N, 32, 0, stream>>>(buf, LD, src, w, c0, cEnd);
    };

    // ---- depth branch: x = chain(depth, l1_1) ----
    append(actA, depth, 1, 0, 32);                      // X = [depth | zeros]
    gemm(actA, LD, 2, actB, LD, 1);                     // l1_1
    gemm(actB, LD, 0, actA, LD, 1);                     // h1
    gemm(actA, LD, 1, actB, LD, 1);                     // h2        -> x in actB
    append(actB, depth, 1, 256, 288);                   // concat depth
    gemm(actB, LD, 3, actA, LD, 1);                     // l1_2
    gemm(actA, LD, 0, actB, LD, 1);                     // h1
    gemm(actB, LD, 1, actA, LD, 1);                     // h2        -> x in actA
    append(actA, fd, 1, 256, 288);                      // concat fd
    gemm(actA, LD, 4, actB, LD, 1);                     // l1_3a
    gemm(actB, LD, 5, actXs, LD, 1);                    // l1_3b     -> x in actXs
    gemm(actXs, LD, 14, actB, LD, 1);                   // sig_a
    gemm(actB, LD, 15, sigB, 16, 0);                    // sig_b     -> sig
    blur_simple_kernel<<<N, 128, 0, stream>>>(sigB, 16, idx, pix, roi, out + 0);

    // ---- x1 branch ----
    append(actXs, imgC, 3, 256, 288);                   // [x, imgC]
    gemm(actXs, LD, 6, actB, LD, 1);                    // l2_1
    gemm(actB, LD, 0, actA, LD, 1);                     // h1
    gemm(actA, LD, 1, actB, LD, 1);                     // h2
    append(actB, imgC, 3, 256, 288);                    // [x1, imgC]
    gemm(actB, LD, 7, actA, LD, 1);                     // l2_2
    gemm(actA, LD, 0, actB, LD, 1);                     // h1
    gemm(actB, LD, 1, actA, LD, 1);                     // h2
    append(actA, rpos, 2, 256, 288);                    // [x1, rays_pos]
    gemm(actA, LD, 8, actB, LD, 1);                     // l2_3a
    gemm(actB, LD, 9, actX1, LD, 1);                    // l2_3b     -> x1 in actX1
    gemm(actX1, LD, 12, actB, LD, 1);                   // mur_a
    gemm(actB, LD, 13, murB, 16, 0);                    // mur_b     -> murho
    paras_kernel<<<(N + 127) / 128, 128, 0, stream>>>(murB, 16, out + (size_t)9 * N, N);
    blur_mu_kernel<<<N, 128, 0, stream>>>(murB, 16, idx, pix, roi, out + (size_t)3 * N);

    // ---- x2 branch ----
    gemm(actX1, LD, 10, actB, LD, 1);                   // l3_1
    gemm(actB, LD, 0, actA, LD, 1);                     // h1
    gemm(actA, LD, 1, actB, LD, 1);                     // h2
    append(actB, imgC, 3, 256, 288);                    // [x2, imgC]
    gemm(actB, LD, 11, actA, LD, 1);                    // l3_2
    gemm(actA, LD, 0, actB, LD, 1);                     // h1
    gemm(actB, LD, 1, actA, LD, 1);                     // h2        -> x2 in actA
    gemm(actA, LD, 16, waB, 512, 1);                    // w_a (512, relu)
    gemm(waB, 512, 17, wB, 1024, 0);                    // w_b (961 -> pad 1024, linear)
    blur_w_kernel<<<N, 128, 0, stream>>>(wB, 1024, idx, pix, roi, out + (size_t)6 * N);
}
